// DeepGCN_11373073400093
// MI455X (gfx1250) — matrix operand order repaired and checked by reading, not compiled, not run on hardware
//
#include <hip/hip_runtime.h>
#include <hip/hip_bf16.h>

// ---------------------------------------------------------------------------
// DeepGCN on MI455X (gfx1250, wave32, WMMA)
//
// Pipeline:
//   1) knn_kernel          : top-20 neighbors per point (LDS-tiled distances)
//   2) per layer:
//        gather_kernel     : build bf16 edge features E = [mean_k(nbr)-ctr, ctr]
//                            (or identity copy for residual 1x1 linears)
//        prepw_kernel      : fp32 W (fin x fout) -> bf16 W^T padded (Npad x Kpad)
//        wmma_gemm_kernel  : [32768 x Kpad] @ [Kpad x Npad] with
//                            v_wmma_f32_16x16x32_bf16; each wave keeps ONE
//                            A-fragment and accumulates NT (<=4) N-tiles,
//                            fused bias / leaky-relu / residual epilogue
//   3) elementwise add for equal-channel residual paths
// ---------------------------------------------------------------------------

#define TOT    32768          // B * N
#define NPTS   8192
#define NBATCH 4
#define KNN    20
#define TILE   256

typedef __attribute__((ext_vector_type(16))) __bf16 v16bf;
typedef __attribute__((ext_vector_type(8)))  __bf16 v8bf;
typedef __attribute__((ext_vector_type(8)))  float  v8f;

// ---------------------------------------------------------------------------
// KNN: one thread per query point, candidates tiled through LDS.
// ---------------------------------------------------------------------------
__global__ void knn_kernel(const float* __restrict__ x, int* __restrict__ idx) {
    const int b = blockIdx.y;
    const int q = blockIdx.x * blockDim.x + threadIdx.x;   // 0..NPTS-1
    const float* xb = x + (size_t)b * NPTS * 4;

    const float qx = xb[q * 4 + 0];
    const float qy = xb[q * 4 + 1];
    const float qz = xb[q * 4 + 2];

    float best[KNN];
    int   bid [KNN];
#pragma unroll
    for (int i = 0; i < KNN; ++i) { best[i] = -3.0e38f; bid[i] = 0; }

    __shared__ float sp[TILE * 3];

    for (int t0 = 0; t0 < NPTS; t0 += TILE) {
        for (int i = threadIdx.x; i < TILE * 3; i += blockDim.x) {
            const int pt = t0 + i / 3;
            const int c  = i % 3;
            sp[i] = xb[pt * 4 + c];
        }
        if (t0 + TILE < NPTS)
            __builtin_prefetch(xb + (size_t)(t0 + TILE) * 4 + threadIdx.x * 4, 0, 1);
        __syncthreads();

        for (int j = 0; j < TILE; ++j) {
            const float dx = sp[j * 3 + 0] - qx;
            const float dy = sp[j * 3 + 1] - qy;
            const float dz = sp[j * 3 + 2] - qz;
            const float nd = -(dx * dx + dy * dy + dz * dz);
            if (nd > best[KNN - 1]) {
                best[KNN - 1] = nd;
                bid [KNN - 1] = b * NPTS + t0 + j;
#pragma unroll
                for (int i = KNN - 1; i > 0; --i) {
                    if (best[i] > best[i - 1]) {        // strict > keeps earlier index on ties
                        const float tv = best[i]; best[i] = best[i - 1]; best[i - 1] = tv;
                        const int   ti = bid [i]; bid [i] = bid [i - 1]; bid [i - 1] = ti;
                    }
                }
            }
        }
        __syncthreads();
    }

    int* out = idx + (size_t)(b * NPTS + q) * KNN;
#pragma unroll
    for (int i = 0; i < KNN; ++i) out[i] = bid[i];
}

// ---------------------------------------------------------------------------
// Gather / edge-feature builder.
// mode 0: E[n, 0:C] = mean_k(nbr) - ctr ; E[n, C:2C] = ctr ; pad 0
// mode 1: E[n, 0:C] = act[n] ; pad 0                (identity for 1x1 linears)
// ---------------------------------------------------------------------------
__global__ void gather_kernel(const float* __restrict__ act,
                              const int*   __restrict__ idx,
                              __bf16*      __restrict__ E,
                              int C, int Kpad, int mode) {
    const long long t = (long long)blockIdx.x * blockDim.x + threadIdx.x;
    if (t >= (long long)TOT * Kpad) return;
    const int n = (int)(t / Kpad);
    const int j = (int)(t - (long long)n * Kpad);

    if (mode == 1) {
        const float v = (j < C) ? act[(size_t)n * C + j] : 0.0f;
        E[t] = (__bf16)v;
        return;
    }

    if (j < C) {
        const float ctr = act[(size_t)n * C + j];
        const int* id = idx + (size_t)n * KNN;
        float s = 0.0f;
#pragma unroll 5
        for (int k = 0; k < KNN; ++k)
            s += act[(size_t)id[k] * C + j];
        const float m = s * (1.0f / (float)KNN);
        E[(size_t)n * Kpad + j]     = (__bf16)(m - ctr);
        E[(size_t)n * Kpad + C + j] = (__bf16)ctr;
    } else if (j >= 2 * C) {
        E[t] = (__bf16)0.0f;
    }
    // j in [C, 2C): slot written by the j-C thread above
}

// ---------------------------------------------------------------------------
// Weight prep: fp32 W (fin x fout, row-major) -> bf16 transposed + padded,
// WT[n * Kpad + k] so each output column's K-run is contiguous for B-fragment
// ---------------------------------------------------------------------------
__global__ void prepw_kernel(const float* __restrict__ W, __bf16* __restrict__ WT,
                             int fin, int fout, int Kpad, int Npad) {
    const int t = blockIdx.x * blockDim.x + threadIdx.x;
    if (t >= Npad * Kpad) return;
    const int n = t / Kpad;
    const int k = t - n * Kpad;
    const float v = (n < fout && k < fin) ? W[(size_t)k * fout + n] : 0.0f;
    WT[t] = (__bf16)v;
}

// ---------------------------------------------------------------------------
// WMMA GEMM with A-fragment reuse across NT adjacent N-tiles per wave.
// Lane layouts per ISA 7.12.2:
//   A 16x32 bf16: lanes 0-15 row=lane, K{k..k+7, k+16..k+23};
//                 lanes 16-31 row=lane-16, K{k+8..k+15, k+24..k+31}
//   B 32x16 bf16: lanes 0-15 col=lane, K{k..k+7, k+16..k+23}; lanes 16-31 K{k+8..k+15, k+24..k+31}
//   C/D f32: VGPR i -> row i (lanes 0-15) / row i+8 (lanes 16-31), col=lane&15
// ---------------------------------------------------------------------------
template <int NT>
__global__ void wmma_gemm_kernel(const __bf16* __restrict__ E,
                                 const __bf16* __restrict__ WT,
                                 const float*  __restrict__ bias,
                                 const float*  __restrict__ res,
                                 float*        __restrict__ out,
                                 int Kpad, int Nout, int ngroups_n, int total_waves,
                                 int do_lrelu, int do_res) {
    const int wave = blockIdx.x * (blockDim.x >> 5) + (threadIdx.x >> 5);
    if (wave >= total_waves) return;               // whole-wave uniform exit
    const int lane = threadIdx.x & 31;
    const int mt = wave / ngroups_n;
    const int ng = wave - mt * ngroups_n;          // group of NT n-tiles

    const int colL = lane & 15;
    const int hi   = lane >> 4;

    const __bf16* Aptr  = E  + (size_t)(mt * 16 + colL) * Kpad + hi * 8;
    const __bf16* Bbase = WT + (size_t)(ng * NT * 16 + colL) * Kpad + hi * 8;

    v8f acc[NT];
#pragma unroll
    for (int t = 0; t < NT; ++t) acc[t] = (v8f){};

    for (int k = 0; k < Kpad; k += 32) {
        union { v16bf v; v8bf h[2]; } a;
        a.h[0] = *(const v8bf*)(Aptr + k);
        a.h[1] = *(const v8bf*)(Aptr + k + 16);
#pragma unroll
        for (int t = 0; t < NT; ++t) {
            union { v16bf v; v8bf h[2]; } b;
            b.h[0] = *(const v8bf*)(Bbase + (size_t)t * 16 * Kpad + k);
            b.h[1] = *(const v8bf*)(Bbase + (size_t)t * 16 * Kpad + k + 16);
#if defined(__HIP_DEVICE_COMPILE__)
            acc[t] = __builtin_amdgcn_wmma_f32_16x16x32_bf16(
                /*neg_a=*/false, a.v, /*neg_b=*/false, b.v,
                /*c_mod=*/(short)0, acc[t], /*reuse_a=*/false, /*reuse_b=*/false);
#else
            (void)b;
#endif
        }
    }

    const int rowbase = mt * 16 + (hi ? 8 : 0);
#pragma unroll
    for (int t = 0; t < NT; ++t) {
        const int col = (ng * NT + t) * 16 + colL;
        if (col < Nout) {
            const float bv = bias[col];
#pragma unroll
            for (int i = 0; i < 8; ++i) {
                const int row = rowbase + i;
                float v = acc[t][i] + bv;
                if (do_lrelu) v = v > 0.0f ? v : 0.2f * v;
                if (do_res)   v += res[(size_t)row * Nout + col];
                out[(size_t)row * Nout + col] = v;
            }
        }
    }
}

__global__ void add_kernel(float* __restrict__ out, const float* __restrict__ a,
                           const float* __restrict__ b, int n) {
    const int i = blockIdx.x * blockDim.x + threadIdx.x;
    if (i < n) out[i] = a[i] + b[i];
}

// ---------------------------------------------------------------------------
// Host orchestration
// ---------------------------------------------------------------------------
struct Lin { const float* W; const float* b; int fin; int fout; };

extern "C" void kernel_launch(void* const* d_in, const int* in_sizes, int n_in,
                              void* d_out, int out_size, void* d_ws, size_t ws_size,
                              hipStream_t stream) {
    (void)in_sizes; (void)n_in; (void)out_size; (void)ws_size;

    const float* x = (const float*)d_in[0];

    // ---- workspace carve (256B aligned) ----
    char* wsp = (char*)d_ws;
    auto carve = [&](size_t bytes) -> void* {
        void* r = (void*)wsp;
        wsp += (bytes + 255) & ~(size_t)255;
        return r;
    };
    int*    idx  = (int*)   carve((size_t)TOT * KNN * sizeof(int));
    float*  bufA = (float*) carve((size_t)TOT * 128 * sizeof(float));
    float*  bufB = (float*) carve((size_t)TOT * 128 * sizeof(float));
    float*  bufU = (float*) carve((size_t)TOT * 128 * sizeof(float));
    float*  bufM = (float*) carve((size_t)TOT * 128 * sizeof(float));
    __bf16* Ebuf = (__bf16*)carve((size_t)TOT * 256 * sizeof(__bf16));
    __bf16* WT   = (__bf16*)carve((size_t)256 * 128 * sizeof(__bf16));

    // ---- parameter walk (setup_inputs insertion order) ----
    int p = 1;
    auto take = [&](int fin, int fout) -> Lin {
        Lin L; L.W = (const float*)d_in[p]; L.b = (const float*)d_in[p + 1];
        L.fin = fin; L.fout = fout; p += 2; return L;
    };

    // ---- launch helpers ----
    auto run_gemm = [&](const __bf16* E, const float* bias, const float* res,
                        float* out, int Kpad, int Nout, int lrelu) {
        const int Npad = (Nout + 15) & ~15;
        const int ntn  = Npad / 16;                  // 1, 2, 4 or 8
        const int NT   = ntn > 4 ? 4 : ntn;          // n-tiles per wave
        const int ngrp = ntn / NT;                   // wave groups along N
        const int total = (TOT / 16) * ngrp;
        const int grid  = (total + 3) / 4;           // 4 waves / block (128 thr)
        const int dres  = res != nullptr ? 1 : 0;
        switch (NT) {
        case 1: wmma_gemm_kernel<1><<<grid, 128, 0, stream>>>(E, WT, bias, res, out,
                    Kpad, Nout, ngrp, total, lrelu, dres); break;
        case 2: wmma_gemm_kernel<2><<<grid, 128, 0, stream>>>(E, WT, bias, res, out,
                    Kpad, Nout, ngrp, total, lrelu, dres); break;
        default: wmma_gemm_kernel<4><<<grid, 128, 0, stream>>>(E, WT, bias, res, out,
                    Kpad, Nout, ngrp, total, lrelu, dres); break;
        }
    };
    auto edge_conv = [&](const float* in, int Cin, Lin L, float* out,
                         int lrelu, const float* res) {
        const int Kpad = ((2 * Cin) + 31) & ~31;
        const long long tg = (long long)TOT * Kpad;
        gather_kernel<<<(int)((tg + 255) / 256), 256, 0, stream>>>(in, idx, Ebuf, Cin, Kpad, 0);
        const int Npad = (L.fout + 15) & ~15;
        const int tw = Npad * Kpad;
        prepw_kernel<<<(tw + 255) / 256, 256, 0, stream>>>(L.W, WT, L.fin, L.fout, Kpad, Npad);
        run_gemm(Ebuf, L.b, res, out, Kpad, L.fout, lrelu);
    };
    auto lin_layer = [&](const float* in, int Cin, Lin L, const float* res, float* out) {
        const int Kpad = (Cin + 31) & ~31;
        const long long tg = (long long)TOT * Kpad;
        gather_kernel<<<(int)((tg + 255) / 256), 256, 0, stream>>>(in, idx, Ebuf, Cin, Kpad, 1);
        const int Npad = (L.fout + 15) & ~15;
        const int tw = Npad * Kpad;
        prepw_kernel<<<(tw + 255) / 256, 256, 0, stream>>>(L.W, WT, L.fin, L.fout, Kpad, Npad);
        run_gemm(Ebuf, L.b, res, out, Kpad, L.fout, 0);
    };

    // ---- 1) KNN ----
    knn_kernel<<<dim3(NPTS / 256, NBATCH), 256, 0, stream>>>(x, idx);

    // ---- 2) entry ----
    Lin entry = take(8, 8);
    edge_conv(x, 4, entry, bufA, 1, nullptr);
    float* cur = bufA; float* alt = bufB; int curC = 8;

    // ---- 3) residual modules ----
    const int nunits[5]  = {2, 3, 2, 3, 2};
    const int chs[5][3]  = {{8,16,16},{32,32,64},{64,64,128},{64,32,32},{16,8,8}};
    int fc = 8;

    for (int mi = 0; mi < 5; ++mi) {
        hipMemcpyAsync(bufM, cur, (size_t)TOT * curC * sizeof(float),
                       hipMemcpyDeviceToDevice, stream);
        const int modC = curC;

        for (int ui = 0; ui < nunits[mi]; ++ui) {
            hipMemcpyAsync(bufU, cur, (size_t)TOT * curC * sizeof(float),
                           hipMemcpyDeviceToDevice, stream);
            const int uC = curC;
            for (int ci = 0; ci < 3; ++ci) {
                const int co = chs[mi][ci];
                Lin L = take(curC * 2, co);
                edge_conv(cur, curC, L, alt, 1, nullptr);
                float* t = cur; cur = alt; alt = t; curC = co;
            }
            Lin f = take(uC, chs[mi][2]);
            if (uC != curC) {
                lin_layer(bufU, uC, f, /*res=*/cur, alt);
            } else {
                const int n = TOT * curC;
                add_kernel<<<(n + 255) / 256, 256, 0, stream>>>(alt, bufU, cur, n);
            }
            { float* t = cur; cur = alt; alt = t; }
        }

        Lin mf = take(fc, chs[mi][2]);
        if (modC != curC) {
            lin_layer(bufM, modC, mf, /*res=*/cur, alt);
        } else {
            const int n = TOT * curC;
            add_kernel<<<(n + 255) / 256, 256, 0, stream>>>(alt, bufM, cur, n);
        }
        { float* t = cur; cur = alt; alt = t; }
        fc = chs[mi][2];
    }

    // ---- 4) exit: edge_conv(fc*2 -> 1), lrelu, straight into d_out ----
    Lin ex = take(fc * 2, 1);
    edge_conv(cur, curC, ex, (float*)d_out, 1, nullptr);
}
